// GnnCl_35192962024016
// MI455X (gfx1250) — matrix operand order repaired and checked by reading, not compiled, not run on hardware
//
#include <hip/hip_runtime.h>
#include <hip/hip_bf16.h>
#include <math.h>

// ---------------- model constants ----------------
#define N_NODE 40000
#define EMB    100
#define BATCH  512
#define SEQL   50
#define NITEM  50
#define NEDGE  800000
#define KPAD   128   // K=100 padded to 128 for 4x WMMA 16x16x32 f16

typedef _Float16 v16h __attribute__((ext_vector_type(16)));
typedef float    v8f  __attribute__((ext_vector_type(8)));

// ---------------- 1) sparse scatter: xout[rows[e]] += vals[e]*xin[cols[e]] ----------------
__global__ __launch_bounds__(256) void spmv_kernel(const int* __restrict__ rows,
                                                   const int* __restrict__ cols,
                                                   const float* __restrict__ vals,
                                                   const float* __restrict__ xin,
                                                   float* __restrict__ xout) {
  long long idx = (long long)blockIdx.x * blockDim.x + threadIdx.x;
  int d = (int)(idx & 127);
  long long e = idx >> 7;
  if (e >= NEDGE || d >= EMB) return;
  int r = rows[e], c = cols[e];
  atomicAdd(&xout[(size_t)r * EMB + d], vals[e] * xin[(size_t)c * EMB + d]);
}

// ---------------- 2) per-row l2norm of 3 layers + weighted sum (wave per row) ----------------
__global__ __launch_bounds__(256) void combine3_kernel(const float* __restrict__ x0,
                                                       const float* __restrict__ x1,
                                                       const float* __restrict__ x2,
                                                       const float* __restrict__ coef,
                                                       float* __restrict__ out, int nrows) {
  int gt   = blockIdx.x * blockDim.x + threadIdx.x;
  int row  = gt >> 5;
  int lane = threadIdx.x & 31;
  if (row >= nrows) return;
  float s0 = 0.f, s1 = 0.f, s2 = 0.f;
  for (int d = lane; d < EMB; d += 32) {
    float v0 = x0[(size_t)row * EMB + d];
    float v1 = x1[(size_t)row * EMB + d];
    float v2 = x2[(size_t)row * EMB + d];
    s0 += v0 * v0; s1 += v1 * v1; s2 += v2 * v2;
  }
  #pragma unroll
  for (int off = 16; off; off >>= 1) {
    s0 += __shfl_xor(s0, off, 32);
    s1 += __shfl_xor(s1, off, 32);
    s2 += __shfl_xor(s2, off, 32);
  }
  float c0 = coef[0] / fmaxf(sqrtf(s0), 1e-12f);
  float c1 = coef[1] / fmaxf(sqrtf(s1), 1e-12f);
  float c2 = coef[2] / fmaxf(sqrtf(s2), 1e-12f);
  for (int d = lane; d < EMB; d += 32) {
    out[(size_t)row * EMB + d] = c0 * x0[(size_t)row * EMB + d]
                               + c1 * x1[(size_t)row * EMB + d]
                               + c2 * x2[(size_t)row * EMB + d];
  }
}

// ---------------- 3) fused GAT per batch: embed lookup + 4 scores + mask + softmax + aggregate ----------------
__global__ __launch_bounds__(256) void gat_kernel(const float* __restrict__ emb,
                                                  const int* __restrict__ items,
                                                  const int* __restrict__ adj,
                                                  const float* __restrict__ a0,
                                                  const float* __restrict__ a1,
                                                  const float* __restrict__ a2,
                                                  const float* __restrict__ a3,
                                                  float* __restrict__ hgat) {
  __shared__ float hS[NITEM * EMB];     // 20 KB
  __shared__ float aS[4 * EMB];         // 1.6 KB
  __shared__ float eS[NITEM * NITEM];   // 10 KB
  int b = blockIdx.x, tid = threadIdx.x;
  for (int idx = tid; idx < NITEM * EMB; idx += 256) {
    int i = idx / EMB, d = idx - i * EMB;
    int it = items[b * NITEM + i];
    hS[idx] = (it == 0) ? 0.f : emb[(size_t)(it - 1) * EMB + d];
  }
  for (int d = tid; d < EMB; d += 256) {
    aS[d] = a0[d]; aS[EMB + d] = a1[d]; aS[2 * EMB + d] = a2[d]; aS[3 * EMB + d] = a3[d];
  }
  __syncthreads();
  for (int p = tid; p < NITEM * NITEM; p += 256) {
    int i = p / NITEM, j = p - i * NITEM;
    const float* hi = &hS[i * EMB];
    const float* hj = &hS[j * EMB];
    float s0 = 0.f, s1 = 0.f, s2 = 0.f, s3 = 0.f;
    for (int d = 0; d < EMB; ++d) {
      float prod = hi[d] * hj[d];
      s0 += prod * aS[d]; s1 += prod * aS[EMB + d];
      s2 += prod * aS[2 * EMB + d]; s3 += prod * aS[3 * EMB + d];
    }
    s0 = (s0 > 0.f) ? s0 : 0.2f * s0;
    s1 = (s1 > 0.f) ? s1 : 0.2f * s1;
    s2 = (s2 > 0.f) ? s2 : 0.2f * s2;
    s3 = (s3 > 0.f) ? s3 : 0.2f * s3;
    int code = adj[(size_t)b * NITEM * NITEM + p];
    float v = -9e15f;
    if      (code == 1) v = s0;
    else if (code == 2) v = s1;
    else if (code == 3) v = s2;
    else if (code == 4) v = s3;
    eS[p] = v;
  }
  __syncthreads();
  // wave32 softmax per row
  int wv = tid >> 5, lane = tid & 31;
  for (int i = wv; i < NITEM; i += 8) {
    float v1 = eS[i * NITEM + lane];                               // lane < 32 < 50
    float v2 = (lane + 32 < NITEM) ? eS[i * NITEM + lane + 32] : -3.4e38f;
    float m = fmaxf(v1, v2);
    #pragma unroll
    for (int off = 16; off; off >>= 1) m = fmaxf(m, __shfl_xor(m, off, 32));
    float e1 = expf(v1 - m);
    float e2 = (lane + 32 < NITEM) ? expf(v2 - m) : 0.f;
    float s = e1 + e2;
    #pragma unroll
    for (int off = 16; off; off >>= 1) s += __shfl_xor(s, off, 32);
    float inv = 1.f / s;
    eS[i * NITEM + lane] = e1 * inv;
    if (lane + 32 < NITEM) eS[i * NITEM + lane + 32] = e2 * inv;
  }
  __syncthreads();
  for (int idx = tid; idx < NITEM * EMB; idx += 256) {
    int i = idx / EMB, d = idx - i * EMB;
    float acc = 0.f;
    for (int j = 0; j < NITEM; ++j) acc += eS[i * NITEM + j] * hS[j * EMB + d];
    hgat[(size_t)b * NITEM * EMB + idx] = acc;
  }
}

// ---------------- 4) get_seq: 0.2*e1t[rev] + 0.8*h[b, alias] ----------------
__global__ __launch_bounds__(256) void getseq_kernel(const float* __restrict__ emb,
                                                     const int* __restrict__ rev,
                                                     const float* __restrict__ hg,
                                                     const int* __restrict__ alias,
                                                     float* __restrict__ seq) {
  int idx = blockIdx.x * blockDim.x + threadIdx.x;
  if (idx >= BATCH * SEQL * EMB) return;
  int d = idx % EMB;
  int bl = idx / EMB;
  int l = bl % SEQL, b = bl / SEQL;
  int r = rev[b * SEQL + l];
  float eh = (r == 0) ? 0.f : emb[(size_t)(r - 1) * EMB + d];
  int al = alias[b * SEQL + l];
  float hh = hg[((size_t)b * NITEM + al) * EMB + d];
  seq[idx] = 0.2f * eh + 0.8f * hh;
}

// ---------------- 5) fused soft attention per batch ----------------
__global__ __launch_bounds__(256) void softatt_kernel(const float* __restrict__ seq,
                                                      const int* __restrict__ mask,
                                                      const float* __restrict__ slen,
                                                      const float* __restrict__ posemb,
                                                      const float* __restrict__ w1W,
                                                      const float* __restrict__ w1b,
                                                      const float* __restrict__ glu1W,
                                                      const float* __restrict__ glu1b,
                                                      const float* __restrict__ glu2W,
                                                      const float* __restrict__ w2,
                                                      float* __restrict__ sel) {
  __shared__ float sS[SEQL * EMB];   // 20 KB
  __shared__ float nhS[SEQL * EMB];  // 20 KB
  __shared__ float hsS[EMB];
  __shared__ float hgS[EMB];
  __shared__ float betaS[SEQL];
  int b = blockIdx.x, tid = threadIdx.x;
  for (int idx = tid; idx < SEQL * EMB; idx += 256) sS[idx] = seq[(size_t)b * SEQL * EMB + idx];
  if (tid < SEQL) betaS[tid] = 0.f;
  __syncthreads();
  if (tid < EMB) {
    float acc = 0.f;
    for (int l = 0; l < SEQL; ++l) acc += sS[l * EMB + tid];
    hsS[tid] = acc / slen[b];
  }
  __syncthreads();
  if (tid < EMB) {
    float acc = 0.f;
    const float* w = &glu2W[tid * EMB];
    for (int k = 0; k < EMB; ++k) acc += hsS[k] * w[k];
    hgS[tid] = acc;
  }
  __syncthreads();
  for (int idx = tid; idx < SEQL * EMB; idx += 256) {
    int l = idx / EMB, d = idx - l * EMB;
    const float* w = &w1W[d * 2 * EMB];
    const float* p = &posemb[l * EMB];
    const float* s = &sS[l * EMB];
    float acc = w1b[d];
    for (int k = 0; k < EMB; ++k) acc += p[k] * w[k] + s[k] * w[EMB + k];
    nhS[idx] = tanhf(acc);
  }
  __syncthreads();
  for (int idx = tid; idx < SEQL * EMB; idx += 256) {
    int l = idx / EMB, d = idx - l * EMB;
    const float* w = &glu1W[d * EMB];
    const float* nh = &nhS[l * EMB];
    float acc = glu1b[d] + hgS[d];
    for (int k = 0; k < EMB; ++k) acc += nh[k] * w[k];
    float sg = 1.f / (1.f + expf(-acc));
    atomicAdd(&betaS[l], sg * w2[d]);
  }
  __syncthreads();
  if (tid < EMB) {
    float acc = 0.f;
    for (int l = 0; l < SEQL; ++l) {
      float bt = betaS[l] * (float)mask[b * SEQL + l];
      acc += bt * sS[l * EMB + tid];
    }
    sel[(size_t)b * EMB + tid] = acc;
  }
}

// ---------------- 6) dense [512x512] @ [512x100] ----------------
__global__ __launch_bounds__(256) void sessmv_kernel(const float* __restrict__ A,
                                                     const float* __restrict__ x,
                                                     float* __restrict__ y) {
  int idx = blockIdx.x * blockDim.x + threadIdx.x;
  if (idx >= BATCH * EMB) return;
  int i = idx / EMB, d = idx - i * EMB;
  const float* arow = &A[(size_t)i * BATCH];
  float acc = 0.f;
  for (int j = 0; j < BATCH; ++j) acc += arow[j] * x[(size_t)j * EMB + d];
  y[idx] = acc;
}

// ---------------- 7) f32 -> f16 with K pad to 128 ----------------
__global__ __launch_bounds__(256) void cvt16_kernel(const float* __restrict__ in,
                                                    _Float16* __restrict__ out, int nrows) {
  int idx = blockIdx.x * blockDim.x + threadIdx.x;
  if (idx >= nrows * KPAD) return;
  int c = idx & (KPAD - 1), r = idx >> 7;
  out[idx] = (c < EMB) ? (_Float16)in[(size_t)r * EMB + c] : (_Float16)0.f;
}

// ---------------- 8) scores GEMM via v_wmma_f32_16x16x32_f16 ----------------
// C[512, 40000] = 4 * A[512,128] @ B[40000,128]^T ; one wave per 16x16 tile, 4 K steps.
__global__ __launch_bounds__(128) void score_gemm_wmma(const _Float16* __restrict__ A,
                                                       const _Float16* __restrict__ B,
                                                       float* __restrict__ C) {
  int wave = threadIdx.x >> 5;
  int lane = threadIdx.x & 31;
  int half = lane >> 4;
  int sub  = lane & 15;
  int nTile = blockIdx.x * 4 + wave;   // 2500 N tiles
  int mTile = blockIdx.y;              // 32 M tiles
  const _Float16* arow = A + (size_t)(mTile * 16 + sub) * KPAD;
  const _Float16* brow = B + (size_t)(nTile * 16 + sub) * KPAD;
  v8f c;
  #pragma unroll
  for (int r = 0; r < 8; ++r) c[r] = 0.f;
  #pragma unroll
  for (int kt = 0; kt < 4; ++kt) {
    v16h a, bb;
    // A 16x32 f16 layout: lane m, elems K=e+16*half
    #pragma unroll
    for (int e = 0; e < 16; ++e) a[e]    = arow[kt * 32 + 16 * half + e];
    // B 32x16 f16 layout: lane col n, elems K=e+16*half (contiguous in emb row)
    #pragma unroll
    for (int e = 0; e < 16; ++e) bb[e]   = brow[kt * 32 + 16 * half + e];
    c = __builtin_amdgcn_wmma_f32_16x16x32_f16(false, a, false, bb, (short)0, c,
                                               false, false);
  }
  float* crow = C + (size_t)(mTile * 16 + 8 * half) * N_NODE + nTile * 16 + sub;
  #pragma unroll
  for (int r = 0; r < 8; ++r) crow[(size_t)r * N_NODE] = 4.0f * c[r];
}

// ---------------- 9) per-row max / sum-exp over 40000 ----------------
__global__ __launch_bounds__(256) void rowred_kernel(const float* __restrict__ S,
                                                     float* __restrict__ rmax,
                                                     float* __restrict__ rsum) {
  __shared__ float red[256];
  int b = blockIdx.x, tid = threadIdx.x;
  const float* row = &S[(size_t)b * N_NODE];
  float m = -3.4e38f;
  for (int j = tid; j < N_NODE; j += 256) m = fmaxf(m, row[j]);
  red[tid] = m; __syncthreads();
  for (int o = 128; o; o >>= 1) { if (tid < o) red[tid] = fmaxf(red[tid], red[tid + o]); __syncthreads(); }
  m = red[0]; __syncthreads();
  float s = 0.f;
  for (int j = tid; j < N_NODE; j += 256) s += expf(row[j] - m);
  red[tid] = s; __syncthreads();
  for (int o = 128; o; o >>= 1) { if (tid < o) red[tid] += red[tid + o]; __syncthreads(); }
  if (tid == 0) { rmax[b] = m; rsum[b] = red[0]; }
}

// ---------------- 10) final loss ----------------
__global__ __launch_bounds__(512) void loss_kernel(const float* __restrict__ S,
                                                   const int* __restrict__ tar,
                                                   const float* __restrict__ rmax,
                                                   const float* __restrict__ rsum,
                                                   float* __restrict__ out) {
  __shared__ float red[512];
  int tid = threadIdx.x;
  int t = tar[tid];
  float lp = S[(size_t)tid * N_NODE + t] - rmax[tid] - logf(rsum[tid]);
  red[tid] = lp; __syncthreads();
  for (int o = 256; o; o >>= 1) { if (tid < o) red[tid] += red[tid + o]; __syncthreads(); }
  if (tid == 0) { out[0] = 0.f; out[1] = -red[0] / (float)BATCH; }
}

// ---------------- launcher ----------------
extern "C" void kernel_launch(void* const* d_in, const int* in_sizes, int n_in,
                              void* d_out, int out_size, void* d_ws, size_t ws_size,
                              hipStream_t stream) {
  (void)in_sizes; (void)n_in; (void)out_size; (void)ws_size;
  const int*   tar    = (const int*)d_in[0];
  const int*   rev    = (const int*)d_in[1];
  const int*   mask   = (const int*)d_in[2];
  const float* slen   = (const float*)d_in[3];
  const float* sadj   = (const float*)d_in[4];
  const int*   items  = (const int*)d_in[6];
  const int*   adj    = (const int*)d_in[7];
  const int*   alias  = (const int*)d_in[8];
  const int*   grows  = (const int*)d_in[9];
  const int*   gcols  = (const int*)d_in[10];
  const float* gvals  = (const float*)d_in[11];
  const float* nemb   = (const float*)d_in[12];
  const float* bitem  = (const float*)d_in[13];
  const float* asess  = (const float*)d_in[14];
  const float* a0     = (const float*)d_in[15];
  const float* a1     = (const float*)d_in[16];
  const float* a2     = (const float*)d_in[17];
  const float* a3     = (const float*)d_in[18];
  const float* w1W    = (const float*)d_in[19];
  const float* w1b    = (const float*)d_in[20];
  const float* w2     = (const float*)d_in[21];
  const float* glu1W  = (const float*)d_in[22];
  const float* glu1b  = (const float*)d_in[23];
  const float* glu2W  = (const float*)d_in[24];
  const float* posemb = (const float*)d_in[25];

  float* out = (float*)d_out;
  float* scores = out + 2;

  char* ws = (char*)d_ws;
  float*    EMBG = (float*)(ws + 0);             // 16,000,000 B
  float*    X1   = (float*)(ws + 16000000);      // 16,000,000 B (later: GAT output)
  float*    X2   = (float*)(ws + 32000000);      // 16,000,000 B (later: seq)
  _Float16* EMBH = (_Float16*)(ws + 48000000);   // 10,240,000 B
  float*    SELA = (float*)(ws + 58240000);      // 204,800 B
  float*    SELB = (float*)(ws + 58444800);
  float*    SELC = (float*)(ws + 58649600);
  float*    SELD = (float*)(ws + 58854400);
  _Float16* SELH = (_Float16*)(ws + 59059200);   // 131,072 B
  float*    RMAX = (float*)(ws + 59190272);
  float*    RSUM = (float*)(ws + 59192320);
  float*    HG   = X1;
  float*    SEQ  = X2;

  // item graph conv: two sparse layers (L2-resident scatter-add)
  hipMemsetAsync(X1, 0, (size_t)N_NODE * EMB * sizeof(float), stream);
  hipMemsetAsync(X2, 0, (size_t)N_NODE * EMB * sizeof(float), stream);
  spmv_kernel<<<(NEDGE * 128) / 256, 256, 0, stream>>>(grows, gcols, gvals, nemb, X1);
  spmv_kernel<<<(NEDGE * 128) / 256, 256, 0, stream>>>(grows, gcols, gvals, X1, X2);
  combine3_kernel<<<(N_NODE * 32 + 255) / 256, 256, 0, stream>>>(nemb, X1, X2, bitem, EMBG, N_NODE);

  // GAT + get_seq (single stream since e1t==e2t)
  gat_kernel<<<BATCH, 256, 0, stream>>>(EMBG, items, adj, a0, a1, a2, a3, HG);
  getseq_kernel<<<(BATCH * SEQL * EMB + 255) / 256, 256, 0, stream>>>(EMBG, rev, HG, alias, SEQ);

  // soft attention + session conv
  softatt_kernel<<<BATCH, 256, 0, stream>>>(SEQ, mask, slen, posemb, w1W, w1b, glu1W, glu1b, glu2W, w2, SELA);
  sessmv_kernel<<<(BATCH * EMB + 255) / 256, 256, 0, stream>>>(sadj, SELA, SELB);
  sessmv_kernel<<<(BATCH * EMB + 255) / 256, 256, 0, stream>>>(sadj, SELB, SELC);
  combine3_kernel<<<(BATCH * 32 + 255) / 256, 256, 0, stream>>>(SELA, SELB, SELC, asess, SELD, BATCH);

  // f16 conversion (K padded to 128) and WMMA scores GEMM with x4 fold-in
  cvt16_kernel<<<(BATCH * KPAD + 255) / 256, 256, 0, stream>>>(SELD, SELH, BATCH);
  cvt16_kernel<<<(N_NODE * KPAD + 255) / 256, 256, 0, stream>>>(EMBG, EMBH, N_NODE);
  {
    dim3 grid(N_NODE / 16 / 4, BATCH / 16);  // (625, 32)
    score_gemm_wmma<<<grid, 128, 0, stream>>>(SELH, EMBH, scores);
  }

  // log_softmax + NLL loss
  rowred_kernel<<<BATCH, 256, 0, stream>>>(scores, RMAX, RSUM);
  loss_kernel<<<1, 512, 0, stream>>>(scores, tar, RMAX, RSUM, out);
}
